// GNN_32796370272850
// MI455X (gfx1250) — compile-verified, matrix operand order checked by reading, not run on hardware
//
#include <hip/hip_runtime.h>
#include <hip/hip_bf16.h>

typedef __attribute__((ext_vector_type(16))) _Float16 v16h;
typedef __attribute__((ext_vector_type(8)))  float    v8f;

#define GNN_N  50000
#define GNN_E  800000
#define GNN_NL 3
#define GNN_NG 16
#define WAVES  4
#define EDGE_BLOCKS 1250
#define TILES_PER_WAVE 10   // 1250 * 4 * 10 * 16 = 800000 edges

// ---------------------------------------------------------------------------
// Fragment loader: 16-bit A-matrix 16x32 layout (ISA 7.12.2).
// lane 0-15 : row m, K = kbase+0..7 (v0..v3), kbase+16..23 (v4..v7)
// lane 16-31: row m, K = kbase+8..15,        kbase+24..31
// B fragments use the same loader on W^T (column-major B).
// ---------------------------------------------------------------------------
__device__ __forceinline__ v16h load_frag16(const _Float16* __restrict__ base,
                                            int stride, int row, int kbase, int hi) {
  const _Float16* p = base + row * stride + kbase + hi * 8;
  v16h f;
#pragma unroll
  for (int i = 0; i < 8; ++i) f[i] = p[i];
#pragma unroll
  for (int i = 0; i < 8; ++i) f[8 + i] = p[16 + i];
  return f;
}

// Async copy of one 16B chunk, global -> LDS (GVS mode: saddr64 + vgpr offset).
// CDNA5 ISA: VDST = LDS byte address, tracked by ASYNCcnt.
__device__ __forceinline__ void async_b128_to_lds(unsigned lofs, unsigned gofs,
                                                  const void* saddr) {
  asm volatile("global_load_async_to_lds_b128 %0, %1, %2"
               :: "v"(lofs), "v"(gofs), "s"(saddr) : "memory");
}

// Issue async staging of one 16-edge A-tile: cols 0..63 = h16[row[e]],
// cols 64..127 = ea16[e]. 8 x b128 per lane-set (4 rows per instruction).
__device__ __forceinline__ void issue_tile_async(const _Float16* __restrict__ h16,
                                                 const _Float16* __restrict__ ea16,
                                                 const int* __restrict__ rowIdx,
                                                 int base, unsigned ldsTile, int lane) {
  const int rsel  = lane >> 3;   // 0..3
  const int chunk = lane & 7;    // 0..7 (16B chunks of a 128B row)
#pragma unroll
  for (int i = 0; i < 4; ++i) {
    const int row = i * 4 + rsel;
    // edge features: contiguous rows of ea16
    async_b128_to_lds(ldsTile + (unsigned)(row * 256 + 128 + chunk * 16),
                      (unsigned)((base + row) * 128 + chunk * 16), ea16);
    // gathered node features: h16[rowIdx[edge]]
    const int r = rowIdx[base + row];
    async_b128_to_lds(ldsTile + (unsigned)(row * 256 + chunk * 16),
                      (unsigned)(r * 128 + chunk * 16), h16);
  }
}

// ---------------------------------------------------------------------------
// Weight conversion: W1s [NL,128,128] (in,out) -> W1T [NL,128(out),128(in)] f16
//                    W2s [NL,128,64]  (in,out) -> W2T [NL,64(out),128(in)]  f16
// ---------------------------------------------------------------------------
__global__ void w16_kernel(const float* __restrict__ W1s, const float* __restrict__ W2s,
                           _Float16* __restrict__ W1T, _Float16* __restrict__ W2T) {
  int i = blockIdx.x * blockDim.x + threadIdx.x;
  if (i < GNN_NL * 128 * 128) {
    int l = i / 16384, r = i % 16384, o = r / 128, in = r % 128;
    W1T[i] = (_Float16)W1s[(l * 128 + in) * 128 + o];
  }
  int j = i - GNN_NL * 128 * 128;
  if (j >= 0 && j < GNN_NL * 64 * 128) {
    int l = j / 8192, r = j % 8192, o = r / 128, in = r % 128;
    W2T[j] = (_Float16)W2s[(l * 128 + in) * 64 + o];
  }
}

// ---------------------------------------------------------------------------
// Initial pooling: h_sum = scatter-add(edge_attr by col), counts, ge[0] sums,
// and ea (f16 working copy).
// ---------------------------------------------------------------------------
__global__ void edge_init_kernel(const float* __restrict__ edge_attr,
                                 const int* __restrict__ col, const int* __restrict__ eb,
                                 _Float16* __restrict__ ea16,
                                 float* __restrict__ hsum, float* __restrict__ hcnt,
                                 float* __restrict__ gesum0, float* __restrict__ gecnt) {
  int i = blockIdx.x * blockDim.x + threadIdx.x;   // exactly E*64 threads
  int e = i >> 6, f = i & 63;
  float v = edge_attr[i];
  ea16[i] = (_Float16)v;
  int c = col[e], g = eb[e];
  atomicAdd(&hsum[c * 64 + f], v);
  atomicAdd(&gesum0[g * 64 + f], v);
  if (f == 0) { atomicAdd(&hcnt[c], 1.0f); atomicAdd(&gecnt[g], 1.0f); }
}

__global__ void node_cnt_kernel(const int* __restrict__ batch, float* __restrict__ gncnt) {
  int i = blockIdx.x * blockDim.x + threadIdx.x;
  if (i < GNN_N) atomicAdd(&gncnt[batch[i]], 1.0f);
}

// h16 = (f16)(hsum / max(cnt,1)); accumulate graph-node pool numerator.
// (h is consumed only as the f16 WMMA A-operand, so store it as f16.)
__global__ void node_mean_kernel(const float* __restrict__ hsum, const float* __restrict__ hcnt,
                                 _Float16* __restrict__ h16, const int* __restrict__ batch,
                                 float* __restrict__ gnsum_l) {
  int i = blockIdx.x * blockDim.x + threadIdx.x;   // exactly N*64 threads
  int n = i >> 6, f = i & 63;
  float v = hsum[i] / fmaxf(hcnt[n], 1.0f);
  h16[i] = (_Float16)v;
  atomicAdd(&gnsum_l[batch[n] * 64 + f], v);
}

// ---------------------------------------------------------------------------
// Fused per-edge MLP layer: ea' = (relu([h[row]|ea] @ W1 + b1)) @ W2 + b2
// Weights staged to LDS once per block; biases register-resident per lane.
// Each wave runs 10 contiguous 16-edge tiles; the A-tile is double-buffered
// in LDS (offset-selected so it stays in the LDS address space -> ds_load)
// and filled by GLOBAL_LOAD_ASYNC_TO_LDS_B128 one tile ahead (ASYNCcnt).
// ---------------------------------------------------------------------------
__global__ __launch_bounds__(128) void edge_mlp_kernel(
    const _Float16* __restrict__ h16, _Float16* __restrict__ ea16,
    const int* __restrict__ rowIdx, const int* __restrict__ colIdx,
    const int* __restrict__ eb,
    const _Float16* __restrict__ W1T, const float* __restrict__ b1,
    const _Float16* __restrict__ W2T, const float* __restrict__ b2,
    float* __restrict__ hsum, float* __restrict__ gesum_l) {
  extern __shared__ char smem[];
  _Float16* sW1 = (_Float16*)smem;               // [128][128]
  _Float16* sW2 = sW1 + 128 * 128;               // [64][128]
  _Float16* sAb = sW2 + 64 * 128;                // [WAVES][2][16*128] double-buffered
  _Float16* sHb = sAb + WAVES * 2 * 16 * 128;    // [WAVES][16*128]
  int* sCol = (int*)(sHb + WAVES * 16 * 128);    // [WAVES][16]
  int* sGb  = sCol + WAVES * 16;                 // [WAVES][16]

  const int tid  = threadIdx.x;
  const int w    = tid >> 5;
  const int lane = tid & 31;
  const int m    = lane & 15;
  const int hi   = lane >> 4;

  // ---- Stage weights (48KB) into LDS, cooperatively, once per block. ----
  {
    const float4* g1 = (const float4*)W1T;
    float4*       l1 = (float4*)sW1;
    for (int i = tid; i < 128 * 128 / 8; i += 128) l1[i] = g1[i];   // 2048 x 16B
    const float4* g2 = (const float4*)W2T;
    float4*       l2 = (float4*)sW2;
    for (int i = tid; i < 64 * 128 / 8; i += 128) l2[i] = g2[i];    // 1024 x 16B
  }
  __syncthreads();

  // Biases are loop-invariant per lane (depend only on nt and m): keep them
  // in registers so the epilogues have no global loads / loadcnt stalls.
  float bias1[8], bias2[4];
#pragma unroll
  for (int nt = 0; nt < 8; ++nt) bias1[nt] = b1[nt * 16 + m];
#pragma unroll
  for (int nt = 0; nt < 4; ++nt) bias2[nt] = b2[nt * 16 + m];

  // Single LDS base per wave; buffers selected by integer offset so the
  // pointer provably stays in the LDS address space.
  _Float16* sAwave = sAb + w * (2 * 16 * 128);
  const unsigned ldsA0 = (unsigned)(uintptr_t)sAwave;     // buffer 0 LDS byte addr
  _Float16* sH = sHb + w * (16 * 128);
  int* myCol = sCol + w * 16;
  int* myGb  = sGb + w * 16;

  const int waveGlobal = blockIdx.x * WAVES + w;
  const int firstBase  = waveGlobal * TILES_PER_WAVE * 16;

  // Prologue: kick off async staging of tile 0 into buffer 0.
  issue_tile_async(h16, ea16, rowIdx, firstBase, ldsA0, lane);

  unsigned aOff = 0;                  // element offset of current buffer (0 or 2048)
#pragma unroll 1
  for (int it = 0; it < TILES_PER_WAVE; ++it) {
    const int base = firstBase + it * 16;

    // Stage tile it+1 into the other buffer while we compute this one.
    if (it + 1 < TILES_PER_WAVE) {
      __builtin_prefetch(rowIdx + base + 32, 0, 1);
      issue_tile_async(h16, ea16, rowIdx, base + 16,
                       ldsA0 + ((aOff ^ 2048u) * 2u), lane);
    }
    if (lane < 16) {
      myCol[lane] = colIdx[base + lane];
      myGb[lane]  = eb[base + lane];
    }
    // Tile-uniform graph id? (edge_attr_batch is sorted -> almost always yes)
    bool sameGraph = true;
    const int gb0 = myGb[0];
#pragma unroll
    for (int i = 1; i < 16; ++i) sameGraph &= (myGb[i] == gb0);

    // Wait for this tile's 8 async b128 loads (in-order; leave next 8 pending).
    if (it + 1 < TILES_PER_WAVE)
      asm volatile("s_wait_asynccnt 0x8" ::: "memory");
    else
      asm volatile("s_wait_asynccnt 0x0" ::: "memory");

    const _Float16* sA = sAwave + aOff;   // LDS-space base + integer offset
    v16h af[4];
#pragma unroll
    for (int kc = 0; kc < 4; ++kc) af[kc] = load_frag16(sA, 128, m, kc * 32, hi);

    // GEMM1: [16,128] @ [128,128] -> relu -> sH (f16)
#pragma unroll
    for (int nt = 0; nt < 8; ++nt) {
      v8f acc = {};
#pragma unroll
      for (int kc = 0; kc < 4; ++kc) {
        v16h bf = load_frag16(sW1, 128, nt * 16 + m, kc * 32, hi);
        acc = __builtin_amdgcn_wmma_f32_16x16x32_f16(false, af[kc], false, bf,
                                                     (short)0, acc, false, false);
      }
#pragma unroll
      for (int r = 0; r < 8; ++r) {
        float v = acc[r] + bias1[nt];
        v = v > 0.0f ? v : 0.0f;
        sH[(r + hi * 8) * 128 + nt * 16 + m] = (_Float16)v;   // D: n=lane, m=r(+8)
      }
    }

    v16h hf[4];
#pragma unroll
    for (int kc = 0; kc < 4; ++kc) hf[kc] = load_frag16(sH, 128, m, kc * 32, hi);

    // GEMM2: [16,128] @ [128,64] -> ea' + scatter numerators
#pragma unroll
    for (int nt = 0; nt < 4; ++nt) {
      v8f acc = {};
#pragma unroll
      for (int kc = 0; kc < 4; ++kc) {
        v16h bf = load_frag16(sW2, 128, nt * 16 + m, kc * 32, hi);
        acc = __builtin_amdgcn_wmma_f32_16x16x32_f16(false, hf[kc], false, bf,
                                                     (short)0, acc, false, false);
      }
      const int n_out = nt * 16 + m;
      float gsum = 0.0f;
#pragma unroll
      for (int r = 0; r < 8; ++r) {
        const int   mm   = r + hi * 8;
        const int   edge = base + mm;
        const float v    = acc[r] + bias2[nt];
        ea16[(size_t)edge * 64 + n_out] = (_Float16)v;
        atomicAdd(&hsum[myCol[mm] * 64 + n_out], v);
        gsum += v;
        if (!sameGraph) atomicAdd(&gesum_l[myGb[mm] * 64 + n_out], v);
      }
      if (sameGraph) atomicAdd(&gesum_l[gb0 * 64 + n_out], gsum);  // 8x fewer atomics
    }

    aOff ^= 2048u;   // swap double buffers
  }
}

// ---------------------------------------------------------------------------
// Divide pools, emit all_node_attr / all_edge_attr (outputs 0,1) + all_attr ws.
// ---------------------------------------------------------------------------
__global__ void finalize_kernel(const float* __restrict__ gnsum, const float* __restrict__ gesum,
                                const float* __restrict__ gncnt, const float* __restrict__ gecnt,
                                float* __restrict__ out, float* __restrict__ all_attr) {
  int i = blockIdx.x * blockDim.x + threadIdx.x;   // 8192 threads
  int part = i >> 12;        // 0 = node pools, 1 = edge pools
  int rem  = i & 4095;
  int g = rem >> 8, c = rem & 255;
  int j = c >> 6,  f = c & 63;
  const float* sum = part ? gesum : gnsum;
  const float* cnt = part ? gecnt : gncnt;
  float v = sum[(j * 16 + g) * 64 + f] / fmaxf(cnt[g], 1.0f);
  out[part * 4096 + g * 256 + c] = v;
  all_attr[g * 512 + part * 256 + c] = v;
}

// Tiny output head: [16,512]->128->128->32 (2 MFLOP; scalar is fine).
__global__ __launch_bounds__(128) void head_kernel(
    const float* __restrict__ attr,
    const float* __restrict__ Wo1, const float* __restrict__ bo1,
    const float* __restrict__ Wo2, const float* __restrict__ bo2,
    const float* __restrict__ Wo3, const float* __restrict__ bo3,
    float* __restrict__ out) {
  __shared__ float s1[16][128];
  __shared__ float s2[16][128];
  const int t = threadIdx.x;
  for (int g = 0; g < 16; ++g) {
    float acc = bo1[t];
    for (int k = 0; k < 512; ++k) acc += attr[g * 512 + k] * Wo1[k * 128 + t];
    s1[g][t] = fmaxf(acc, 0.0f);
  }
  __syncthreads();
  for (int g = 0; g < 16; ++g) {
    float acc = bo2[t];
    for (int k = 0; k < 128; ++k) acc += s1[g][k] * Wo2[k * 128 + t];
    s2[g][t] = fmaxf(acc, 0.0f);
  }
  __syncthreads();
  if (t < 32) {
    for (int g = 0; g < 16; ++g) {
      float acc = bo3[t];
      for (int k = 0; k < 128; ++k) acc += s2[g][k] * Wo3[k * 32 + t];
      out[g * 32 + t] = acc;
    }
  }
}

// ---------------------------------------------------------------------------
extern "C" void kernel_launch(void* const* d_in, const int* in_sizes, int n_in,
                              void* d_out, int out_size, void* d_ws, size_t ws_size,
                              hipStream_t stream) {
  (void)in_sizes; (void)n_in; (void)out_size; (void)ws_size;

  const float* edge_attr  = (const float*)d_in[1];
  const int*   edge_index = (const int*)d_in[3];
  const int*   batch      = (const int*)d_in[4];
  const int*   ebatch     = (const int*)d_in[5];
  const float* W1s = (const float*)d_in[6];
  const float* b1s = (const float*)d_in[7];
  const float* W2s = (const float*)d_in[8];
  const float* b2s = (const float*)d_in[9];
  const float* Wo1 = (const float*)d_in[10];
  const float* bo1 = (const float*)d_in[11];
  const float* Wo2 = (const float*)d_in[12];
  const float* bo2 = (const float*)d_in[13];
  const float* Wo3 = (const float*)d_in[14];
  const float* bo3 = (const float*)d_in[15];
  float* out = (float*)d_out;

  const int* rowIdx = edge_index;
  const int* colIdx = edge_index + GNN_E;

  // Workspace carve (256B aligned regions; accumulators contiguous for one memset).
  char* p = (char*)d_ws;
  auto carve = [&](size_t bytes) { void* r = (void*)p; p += (bytes + 255) & ~(size_t)255; return r; };
  _Float16* ea16  = (_Float16*)carve((size_t)GNN_E * 64 * sizeof(_Float16));
  _Float16* h16   = (_Float16*)carve((size_t)GNN_N * 64 * sizeof(_Float16));
  char* accBase   = p;
  float*    hsum  = (float*)carve((size_t)GNN_N * 64 * sizeof(float));
  float*    hcnt  = (float*)carve((size_t)GNN_N * sizeof(float));
  float*    gnsum = (float*)carve((size_t)(GNN_NL + 1) * GNN_NG * 64 * sizeof(float));
  float*    gesum = (float*)carve((size_t)(GNN_NL + 1) * GNN_NG * 64 * sizeof(float));
  float*    gncnt = (float*)carve((size_t)GNN_NG * sizeof(float));
  float*    gecnt = (float*)carve((size_t)GNN_NG * sizeof(float));
  size_t accBytes = (size_t)(p - accBase);
  _Float16* W1T   = (_Float16*)carve((size_t)GNN_NL * 128 * 128 * sizeof(_Float16));
  _Float16* W2T   = (_Float16*)carve((size_t)GNN_NL * 64 * 128 * sizeof(_Float16));
  float* all_attr = (float*)carve((size_t)GNN_NG * 512 * sizeof(float));

  hipMemsetAsync(accBase, 0, accBytes, stream);
  w16_kernel<<<(GNN_NL * 128 * 128 + GNN_NL * 64 * 128 + 255) / 256, 256, 0, stream>>>(
      W1s, W2s, W1T, W2T);

  edge_init_kernel<<<GNN_E * 64 / 256, 256, 0, stream>>>(edge_attr, colIdx, ebatch,
                                                         ea16, hsum, hcnt, gesum, gecnt);
  node_cnt_kernel<<<(GNN_N + 255) / 256, 256, 0, stream>>>(batch, gncnt);
  node_mean_kernel<<<GNN_N * 64 / 256, 256, 0, stream>>>(hsum, hcnt, h16, batch, gnsum);

  // Dynamic LDS: W1T(32KB) + W2T(16KB) + A tiles (2*4KB*4) + hid (4KB*4) + meta
  const size_t smemBytes =
      (size_t)(128 * 128 + 64 * 128 + 2 * WAVES * 16 * 128 + WAVES * 16 * 128) * sizeof(_Float16)
      + (size_t)2 * WAVES * 16 * sizeof(int);
  for (int l = 0; l < GNN_NL; ++l) {
    hipMemsetAsync(hsum, 0, (size_t)GNN_N * 64 * sizeof(float), stream);
    edge_mlp_kernel<<<EDGE_BLOCKS, 128, smemBytes, stream>>>(
        h16, ea16, rowIdx, colIdx, ebatch,
        W1T + (size_t)l * 128 * 128, b1s + l * 128,
        W2T + (size_t)l * 64 * 128,  b2s + l * 64,
        hsum, gesum + (size_t)(l + 1) * GNN_NG * 64);
    node_mean_kernel<<<GNN_N * 64 / 256, 256, 0, stream>>>(
        hsum, hcnt, h16, batch, gnsum + (size_t)(l + 1) * GNN_NG * 64);
  }

  finalize_kernel<<<8192 / 256, 256, 0, stream>>>(gnsum, gesum, gncnt, gecnt, out, all_attr);
  head_kernel<<<1, 128, 0, stream>>>(all_attr, Wo1, bo1, Wo2, bo2, Wo3, bo3, out + 8192);
}
